// FeedForward_MoE_13606456394124
// MI455X (gfx1250) — compile-verified, matrix operand order checked by reading, not run on hardware
//
#include <hip/hip_runtime.h>

// Problem constants (match reference)
#define NTOK   16384      // B*T
#define CDIM   1024
#define EXP    8
#define TOPK   2
#define CAPN   4096
#define HDIM   4096

typedef __attribute__((ext_vector_type(16))) __bf16 v16bf;
typedef __attribute__((ext_vector_type(8)))  __bf16 v8bf;
typedef __attribute__((ext_vector_type(8)))  float  v8f;
typedef __attribute__((ext_vector_type(4))) unsigned uint32x4;
typedef __attribute__((ext_vector_type(8))) int      int32x8;
typedef __attribute__((ext_vector_type(4))) int      int32x4;

// Use the Tensor Data Mover when the toolchain exposes the builtin.
// This toolchain uses the 6-arg form: (u32x4, i32x8, i32x4, i32x4, i32x8, cpol).
// Falls back to VMEM staging otherwise (also covers the host pass).
#if defined(__has_builtin)
#if __has_builtin(__builtin_amdgcn_tensor_load_to_lds) && \
    __has_builtin(__builtin_amdgcn_s_wait_tensorcnt)
#define USE_TDM 1
#else
#define USE_TDM 0
#endif
#else
#define USE_TDM 0
#endif

__device__ __forceinline__ unsigned short f2bf_bits(float f) {
    union { float f; unsigned u; } a; a.f = f;
    unsigned u = a.u;
    return (unsigned short)((u + 0x7FFFu + ((u >> 16) & 1u)) >> 16);  // RNE
}
__device__ __forceinline__ float bf2f_bits(unsigned short s) {
    union { unsigned u; float f; } o; o.u = ((unsigned)s) << 16; return o.f;
}

// ---------------------------------------------------------------------------
// f32 -> bf16 elementwise conversion
// ---------------------------------------------------------------------------
__global__ void conv_f32_bf16(const float* __restrict__ in,
                              unsigned short* __restrict__ out, long long n) {
    long long i = (long long)blockIdx.x * blockDim.x + threadIdx.x;
    long long stride = (long long)gridDim.x * blockDim.x;
    for (; i < n; i += stride) out[i] = f2bf_bits(in[i]);
}

__global__ void zero_u32(unsigned* __restrict__ p, long long n) {
    long long i = (long long)blockIdx.x * blockDim.x + threadIdx.x;
    long long stride = (long long)gridDim.x * blockDim.x;
    for (; i < n; i += stride) p[i] = 0u;
}

// ---------------------------------------------------------------------------
// Tiled transpose + f32->bf16: in (R x Cc, row-major) -> out (Cc x R)
// One pass over the weights; amortized across the whole GEMM.
// ---------------------------------------------------------------------------
__global__ __launch_bounds__(256) void transpose_conv(
    const float* __restrict__ in, unsigned short* __restrict__ out,
    int R, int Cc, long long sIn, long long sOut) {
    __shared__ unsigned short tile[32][33];
    const int b = blockIdx.z;
    const float* ip = in + (long long)b * sIn;
    unsigned short* op = out + (long long)b * sOut;
    const int c0 = blockIdx.x * 32, r0 = blockIdx.y * 32;
    const int tx = threadIdx.x & 31, ty = threadIdx.x >> 5;  // 32 x 8
#pragma unroll
    for (int i = 0; i < 32; i += 8)
        tile[ty + i][tx] = f2bf_bits(ip[(long long)(r0 + ty + i) * Cc + c0 + tx]);
    __syncthreads();
#pragma unroll
    for (int i = 0; i < 32; i += 8)
        op[(long long)(c0 + ty + i) * R + r0 + tx] = tile[tx][ty + i];
}

// ---------------------------------------------------------------------------
// TDM: 2D tile load, 128 rows x 32 bf16, row stride = strideElems.
// D# built per ISA 8.3/8.4: group0 = {count|flags, lds_addr, global_addr,
// addr_hi|type=2}; group1 = {data_size=2B, tensor_dim0/1 = tile dims (no OOB),
// tile_dim0=32, tile_dim1=128, tensor_dim0_stride=strideElems}.
// ---------------------------------------------------------------------------
#if USE_TDM
__device__ __forceinline__ void tdm_load_tile(unsigned lds_off, const void* gptr,
                                              unsigned long long strideElems) {
    unsigned long long ga = (unsigned long long)gptr;
    uint32x4 g0;
    g0.x = 1u;                                       // count=1, user mode
    g0.y = lds_off;                                  // LDS byte offset
    g0.z = (unsigned)ga;                             // global_addr[31:0]
    g0.w = (unsigned)((ga >> 32) & 0x01FFFFFFull)    // global_addr[56:32]
           | (2u << 30);                             // type = 2 ("image")
    int32x8 g1;
    g1[0] = 0x00010000;                              // wg_mask=0, data_size=1(2B)
    g1[1] = (int)(32u << 16);                        // tensor_dim0 = 32
    g1[2] = (int)(128u << 16);                       // tensor_dim1 = 128
    g1[3] = (int)(32u << 16);                        // tile_dim0 = 32
    g1[4] = 128;                                     // tile_dim1=128, tile_dim2=0
    g1[5] = (int)(unsigned)(strideElems & 0xFFFFFFFFull);   // stride[31:0]
    g1[6] = (int)(unsigned)((strideElems >> 32) & 0xFFFFull); // stride[47:32]
    g1[7] = 0;                                       // tensor_dim1_stride = 0 (2D)
    int32x4 z4 = {0, 0, 0, 0};
    int32x8 z8 = {0, 0, 0, 0, 0, 0, 0, 0};
    __builtin_amdgcn_tensor_load_to_lds(g0, g1, z4, z4, z8, 0);
}
#endif

// VMEM fallback staging: 256 threads copy a 128x32 bf16 tile (16B x2 each).
__device__ __forceinline__ void stage_tile(const __bf16* __restrict__ src, int ld,
                                           __bf16* __restrict__ dst, int tid) {
    const int row = tid >> 1;
    const int col = (tid & 1) * 16;
    const __bf16* g = src + (long long)row * ld + col;
    *(v8bf*)(dst + row * 32 + col)     = *(const v8bf*)(g);
    *(v8bf*)(dst + row * 32 + col + 8) = *(const v8bf*)(g + 8);
}

// ---------------------------------------------------------------------------
// Fragment load from LDS (row-major tile, ld=32).
// ISA 16-bit operand layout: lane (half*16+r): row r, K = {half*8..+7, 16+half*8..+7}
// -> two 16B ds_load_b128 per fragment.
// ---------------------------------------------------------------------------
__device__ __forceinline__ v16bf ld_frag(const __bf16* base, int row, int half) {
    union { v16bf v; v8bf h[2]; } u;
    const __bf16* p = base + row * 32 + half * 8;
    u.h[0] = *(const v8bf*)(p);
    u.h[1] = *(const v8bf*)(p + 16);
    return u.v;
}

// ---------------------------------------------------------------------------
// Batched bf16 WMMA GEMM, B pre-transposed:
//   C[b] = act(A[b] (MxK, row-major) * Bt[b]^T (Bt: NxK, row-major) + bias[b])
// BM=BN=128, BK=32; 256 threads = 8 waves, each wave 32(M) x 64(N) = 2x4 WMMA.
// Double-buffered LDS; tiles staged by TDM (tensor_load_to_lds) when available.
// Requires M%128==0, N%128==0, K%32==0.
// ---------------------------------------------------------------------------
template <bool RELU, bool OUT_BF16>
__global__ __launch_bounds__(256) void wmma_gemm_bf16(
    const unsigned short* __restrict__ Ag,
    const unsigned short* __restrict__ Btg,
    const float* __restrict__ biasg,
    void* __restrict__ Cg,
    int M, int N, int Kd,
    long long sA, long long sB, long long sBias, long long sC) {
    constexpr int BM = 128, BN = 128, BK = 32;
    __shared__ __align__(16) __bf16 As[2][BM * BK];
    __shared__ __align__(16) __bf16 Bs[2][BN * BK];

    const int batch = blockIdx.z;
    const __bf16* A  = (const __bf16*)Ag  + (long long)batch * sA;
    const __bf16* Bt = (const __bf16*)Btg + (long long)batch * sB;
    const float* bp  = biasg + (long long)batch * sBias;

    const int bm = blockIdx.y * BM;
    const int bn = blockIdx.x * BN;
    const int tid = threadIdx.x;
    const int lane = tid & 31, wave = tid >> 5;
    const int half = lane >> 4, r = lane & 15;
    const int wm = (wave & 3) * 32;   // 4 waves tile M
    const int wn = (wave >> 2) * 64;  // 2 waves tile N

    v8f acc[2][4];
#pragma unroll
    for (int i = 0; i < 2; i++)
#pragma unroll
        for (int j = 0; j < 4; j++) {
            v8f z = {0.f, 0.f, 0.f, 0.f, 0.f, 0.f, 0.f, 0.f};
            acc[i][j] = z;
        }

    const __bf16* Atile0 = A  + (long long)bm * Kd;
    const __bf16* Btile0 = Bt + (long long)bn * Kd;
    const int nk = Kd / BK;

#if USE_TDM
    const unsigned ldsA[2] = {(unsigned)(size_t)&As[0][0], (unsigned)(size_t)&As[1][0]};
    const unsigned ldsB[2] = {(unsigned)(size_t)&Bs[0][0], (unsigned)(size_t)&Bs[1][0]};
    if (wave == 0) {
        tdm_load_tile(ldsA[0], Atile0, (unsigned long long)Kd);
        tdm_load_tile(ldsB[0], Btile0, (unsigned long long)Kd);
    }
#else
    stage_tile(Atile0, Kd, &As[0][0], tid);
    stage_tile(Btile0, Kd, &Bs[0][0], tid);
#endif

    for (int t = 0; t < nk; ++t) {
        const int cur = t & 1;
#if USE_TDM
        if (wave == 0) __builtin_amdgcn_s_wait_tensorcnt(0);  // tile t landed
#endif
        __syncthreads();
        if (t + 1 < nk) {  // stage tile t+1 into the other buffer
#if USE_TDM
            if (wave == 0) {
                tdm_load_tile(ldsA[cur ^ 1], Atile0 + (t + 1) * BK, (unsigned long long)Kd);
                tdm_load_tile(ldsB[cur ^ 1], Btile0 + (t + 1) * BK, (unsigned long long)Kd);
            }
#else
            stage_tile(Atile0 + (t + 1) * BK, Kd, &As[cur ^ 1][0], tid);
            stage_tile(Btile0 + (t + 1) * BK, Kd, &Bs[cur ^ 1][0], tid);
#endif
        }

        v16bf afrag[2], bfrag[4];
#pragma unroll
        for (int i = 0; i < 2; i++) afrag[i] = ld_frag(&As[cur][0], wm + i * 16 + r, half);
#pragma unroll
        for (int j = 0; j < 4; j++) bfrag[j] = ld_frag(&Bs[cur][0], wn + j * 16 + r, half);

#pragma unroll
        for (int i = 0; i < 2; i++)
#pragma unroll
            for (int j = 0; j < 4; j++)
                acc[i][j] = __builtin_amdgcn_wmma_f32_16x16x32_bf16(
                    false, afrag[i], false, bfrag[j], (short)0, acc[i][j],
                    false, false);
    }

    // Epilogue: C/D layout: lane(half,r), VGPR v -> row = v + half*8, col = r
    const long long cbase = (long long)batch * sC;
#pragma unroll
    for (int j = 0; j < 4; j++) {
        const int n = bn + wn + j * 16 + r;
        const float bv = bp[n];
#pragma unroll
        for (int i = 0; i < 2; i++) {
#pragma unroll
            for (int v = 0; v < 8; v++) {
                const int m = bm + wm + i * 16 + half * 8 + v;
                float val = acc[i][j][v] + bv;
                if (RELU) val = fmaxf(val, 0.0f);
                if constexpr (OUT_BF16)
                    ((unsigned short*)Cg)[cbase + (long long)m * N + n] = f2bf_bits(val);
                else
                    ((float*)Cg)[cbase + (long long)m * N + n] = val;
            }
        }
    }
}

// ---------------------------------------------------------------------------
// Router GEMM2: logits[n][e] = sum_k h1[n][k]*rw2[k][e] + rb2[e]  (8 out cols)
// ---------------------------------------------------------------------------
__global__ void router_gemm2(const unsigned short* __restrict__ h1,
                             const float* __restrict__ rw2,
                             const float* __restrict__ rb2,
                             float* __restrict__ logits) {
    const int idx = blockIdx.x * blockDim.x + threadIdx.x;  // n*8 + e
    if (idx >= NTOK * EXP) return;
    const int n = idx >> 3, e = idx & 7;
    const unsigned short* hp = h1 + (long long)n * HDIM;
    float s = rb2[e];
#pragma unroll 4
    for (int k = 0; k < HDIM; k++) s += bf2f_bits(hp[k]) * rw2[k * EXP + e];
    logits[idx] = s;
}

// ---------------------------------------------------------------------------
// Top-2 + softmax (stable tie-break: lower index wins, like lax.top_k)
// ---------------------------------------------------------------------------
__global__ void topk_softmax(const float* __restrict__ logits,
                             int* __restrict__ ids, float* __restrict__ probs) {
    const int n = blockIdx.x * blockDim.x + threadIdx.x;
    if (n >= NTOK) return;
    float v[EXP];
#pragma unroll
    for (int e = 0; e < EXP; e++) v[e] = logits[n * EXP + e];
    int i0 = 0; float m0 = v[0];
#pragma unroll
    for (int e = 1; e < EXP; e++) if (v[e] > m0) { m0 = v[e]; i0 = e; }
    int i1 = -1; float m1 = -3.4e38f;
#pragma unroll
    for (int e = 0; e < EXP; e++)
        if (e != i0 && v[e] > m1) { m1 = v[e]; i1 = e; }
    const float t = expf(m1 - m0);
    const float p0 = 1.0f / (1.0f + t);
    ids[n * 2 + 0] = i0;  ids[n * 2 + 1] = i1;
    probs[n * 2 + 0] = p0; probs[n * 2 + 1] = 1.0f - p0;
}

// ---------------------------------------------------------------------------
// Deterministic per-expert capacity scan (sequential cumsum order preserved)
// ---------------------------------------------------------------------------
__global__ void moe_scan(const int* __restrict__ ids_f, int* __restrict__ pos,
                         int* __restrict__ keep) {
    __shared__ int eid_sh[256];
    __shared__ int base[EXP];
    const int tid = threadIdx.x;
    if (tid < EXP) base[tid] = 0;
    __syncthreads();
    const int total = NTOK * TOPK;  // divisible by 256
    for (int start = 0; start < total; start += 256) {
        const int slot = start + tid;
        const int e = ids_f[slot];
        eid_sh[tid] = e;
        __syncthreads();
        int rank = 0;
        for (int j = 0; j < tid; j++) rank += (eid_sh[j] == e);
        const int p = base[e] + rank;
        pos[slot] = (p < CAPN) ? p : (CAPN - 1);
        keep[slot] = (p < CAPN) ? 1 : 0;
        __syncthreads();
        if (tid < EXP) {
            int c = 0;
            for (int j = 0; j < 256; j++) c += (eid_sh[j] == tid);
            base[tid] += c;
        }
        __syncthreads();
    }
}

// ---------------------------------------------------------------------------
// Scatter kept tokens into per-expert capacity buffers (f32 -> bf16)
// ---------------------------------------------------------------------------
__global__ void scatter_tokens(const float* __restrict__ x,
                               const int* __restrict__ ids_f,
                               const int* __restrict__ pos,
                               const int* __restrict__ keep,
                               unsigned short* __restrict__ disp) {
    const int slot = blockIdx.x;
    if (!keep[slot]) return;
    const int ntok = slot >> 1;
    const int e = ids_f[slot], p = pos[slot];
    const float* src = x + (long long)ntok * CDIM;
    unsigned short* dst = disp + ((long long)e * CAPN + p) * CDIM;
    for (int c = threadIdx.x; c < CDIM; c += blockDim.x) dst[c] = f2bf_bits(src[c]);
}

// ---------------------------------------------------------------------------
// Combine: y[n] = sum over the 2 slots of prob*keep * eo[e, pos, :]
// ---------------------------------------------------------------------------
__global__ void combine_out(const float* __restrict__ eo,
                            const int* __restrict__ ids_f,
                            const int* __restrict__ pos,
                            const int* __restrict__ keep,
                            const float* __restrict__ probs,
                            float* __restrict__ y) {
    const int n = blockIdx.x;
    const int s0 = 2 * n, s1 = 2 * n + 1;
    const float p0 = keep[s0] ? probs[s0] : 0.0f;
    const float p1 = keep[s1] ? probs[s1] : 0.0f;
    const float* r0 = eo + ((long long)ids_f[s0] * CAPN + pos[s0]) * CDIM;
    const float* r1 = eo + ((long long)ids_f[s1] * CAPN + pos[s1]) * CDIM;
    float* yp = y + (long long)n * CDIM;
    for (int c = threadIdx.x; c < CDIM; c += blockDim.x)
        yp[c] = p0 * r0[c] + p1 * r1[c];
}

// ---------------------------------------------------------------------------
extern "C" void kernel_launch(void* const* d_in, const int* in_sizes, int n_in,
                              void* d_out, int out_size, void* d_ws, size_t ws_size,
                              hipStream_t stream) {
    (void)in_sizes; (void)n_in; (void)out_size; (void)ws_size;
    const float* x   = (const float*)d_in[0];
    const float* rw1 = (const float*)d_in[1];
    const float* rb1 = (const float*)d_in[2];
    const float* rw2 = (const float*)d_in[3];
    const float* rb2 = (const float*)d_in[4];
    const float* ew1 = (const float*)d_in[5];
    const float* eb1 = (const float*)d_in[6];
    const float* ew2 = (const float*)d_in[7];
    const float* eb2 = (const float*)d_in[8];
    float* y = (float*)d_out;

    char* ws = (char*)d_ws;
    size_t off = 0;
    auto wsalloc = [&](size_t bytes) -> void* {
        void* p = ws + off;
        off = (off + bytes + 255) & ~(size_t)255;
        return p;
    };
    unsigned short* x_bf   = (unsigned short*)wsalloc((size_t)NTOK * CDIM * 2);
    unsigned short* rw1t   = (unsigned short*)wsalloc((size_t)CDIM * HDIM * 2);       // H x C
    unsigned short* ew1t   = (unsigned short*)wsalloc((size_t)EXP * CDIM * HDIM * 2); // e: H x C
    unsigned short* ew2t   = (unsigned short*)wsalloc((size_t)EXP * HDIM * CDIM * 2); // e: C x H
    unsigned short* h1r    = (unsigned short*)wsalloc((size_t)NTOK * HDIM * 2);
    float* logits          = (float*)wsalloc((size_t)NTOK * EXP * 4);
    int*   ids             = (int*)wsalloc((size_t)NTOK * TOPK * 4);
    float* probs           = (float*)wsalloc((size_t)NTOK * TOPK * 4);
    int*   pos             = (int*)wsalloc((size_t)NTOK * TOPK * 4);
    int*   keep            = (int*)wsalloc((size_t)NTOK * TOPK * 4);
    unsigned short* disp   = (unsigned short*)wsalloc((size_t)EXP * CAPN * CDIM * 2);
    unsigned short* hexp   = (unsigned short*)wsalloc((size_t)EXP * CAPN * HDIM * 2);
    float* eo              = (float*)wsalloc((size_t)EXP * CAPN * CDIM * 4);

    const dim3 blk(256);

    // 1) convert activations; transpose+convert all weights (B matrices N-major)
    conv_f32_bf16<<<2048, blk, 0, stream>>>(x, x_bf, (long long)NTOK * CDIM);
    transpose_conv<<<dim3(HDIM / 32, CDIM / 32, 1), blk, 0, stream>>>(
        rw1, rw1t, CDIM, HDIM, 0, 0);
    transpose_conv<<<dim3(HDIM / 32, CDIM / 32, EXP), blk, 0, stream>>>(
        ew1, ew1t, CDIM, HDIM, (long long)CDIM * HDIM, (long long)CDIM * HDIM);
    transpose_conv<<<dim3(CDIM / 32, HDIM / 32, EXP), blk, 0, stream>>>(
        ew2, ew2t, HDIM, CDIM, (long long)HDIM * CDIM, (long long)HDIM * CDIM);

    // 2) router GEMM1: relu(x @ rw1 + rb1) -> h1r (bf16)   [16384 x 4096, K=1024]
    wmma_gemm_bf16<true, true><<<dim3(HDIM / 128, NTOK / 128, 1), blk, 0, stream>>>(
        x_bf, rw1t, rb1, h1r, NTOK, HDIM, CDIM, 0, 0, 0, 0);

    // 3) router GEMM2 (8 out cols) + top-2 softmax
    router_gemm2<<<(NTOK * EXP + 255) / 256, blk, 0, stream>>>(h1r, rw2, rb2, logits);
    topk_softmax<<<(NTOK + 255) / 256, blk, 0, stream>>>(logits, ids, probs);

    // 4) capacity scan (deterministic sequential order) + dispatch
    zero_u32<<<4096, blk, 0, stream>>>((unsigned*)disp,
                                       (long long)EXP * CAPN * CDIM / 2);
    moe_scan<<<1, blk, 0, stream>>>(ids, pos, keep);
    scatter_tokens<<<NTOK * TOPK, blk, 0, stream>>>(x, ids, pos, keep, disp);

    // 5) expert FFN (batched over E)
    wmma_gemm_bf16<true, true><<<dim3(HDIM / 128, CAPN / 128, EXP), blk, 0, stream>>>(
        disp, ew1t, eb1, hexp, CAPN, HDIM, CDIM,
        (long long)CAPN * CDIM, (long long)CDIM * HDIM, HDIM, (long long)CAPN * HDIM);
    wmma_gemm_bf16<false, false><<<dim3(CDIM / 128, CAPN / 128, EXP), blk, 0, stream>>>(
        hexp, ew2t, eb2, eo, CAPN, CDIM, HDIM,
        (long long)CAPN * HDIM, (long long)HDIM * CDIM, CDIM, (long long)CAPN * CDIM);

    // 6) gather + gate-weighted combine
    combine_out<<<NTOK, blk, 0, stream>>>(eo, ids, pos, keep, probs, y);
}
